// QBertSelfAttention_6004364280065
// MI455X (gfx1250) — matrix operand order repaired and checked by reading, compiled, not run on hardware
//
#include <hip/hip_runtime.h>

typedef __attribute__((ext_vector_type(16))) __bf16 bf16x16;
typedef __attribute__((ext_vector_type(8)))  float  floatx8;

union FragA { bf16x16 v; unsigned u[8]; };

static constexpr int Bb = 8, Ss = 512, Dd = 1024, Hh = 16, HDd = 64;

// float -> bf16 bits, round-to-nearest-even (pure integer, no __bf16 scalar ops)
__device__ __forceinline__ unsigned short f2bf(float f) {
    unsigned x = __builtin_bit_cast(unsigned, f);
    unsigned r = (x + 0x7fffu + ((x >> 16) & 1u)) >> 16;
    return (unsigned short)r;
}
__device__ __forceinline__ unsigned pack2bf(float lo, float hi) {
    return ((unsigned)f2bf(hi) << 16) | (unsigned)f2bf(lo);
}
__device__ __forceinline__ float hmax16(float v) {
    #pragma unroll
    for (int off = 1; off < 16; off <<= 1) v = fmaxf(v, __shfl_xor(v, off, 32));
    return v;
}
__device__ __forceinline__ float hsum16(float v) {
    #pragma unroll
    for (int off = 1; off < 16; off <<= 1) v += __shfl_xor(v, off, 32);
    return v;
}

// Async copy 16B global -> LDS (gfx1250 GLOBAL_LOAD_ASYNC_TO_LDS_B128, ASYNCcnt).
// Low 32 bits of a flat shared pointer are the LDS byte offset (aperture mapping).
__device__ __forceinline__ void async_copy_b128(void* lds_dst, const void* gsrc) {
    unsigned      loff = (unsigned)(unsigned long long)lds_dst;
    unsigned long long g = (unsigned long long)gsrc;
    asm volatile("global_load_async_to_lds_b128 %0, %1, off"
                 :: "v"(loff), "v"(g) : "memory");
}
__device__ __forceinline__ void wait_async() {
    asm volatile("s_wait_asynccnt 0" ::: "memory");
}

// ---------------------------------------------------------------------------
// Kernel 1: fused QKV projection.  Y = X @ W^T + b for W in {Wq,Wk,Wv}.
// X: [4096,1024] f32, W: [1024,1024] f32 (reduction along fast axis of both).
// Output: q/k/v scratch, bf16, layout [B][H][S][HD].
// Workgroup: 256 threads (8 waves); tile M=128, N=128, K-step 32.
// Wave grid 2(M)x4(N): each wave computes 64(M) x 32(N) = 8 accumulators,
// 8 WMMA per K-step against 6 fragment loads.
// ---------------------------------------------------------------------------
__global__ __launch_bounds__(256) void qkv_gemm_kernel(
    const float* __restrict__ X,
    const float* __restrict__ Wq, const float* __restrict__ bq,
    const float* __restrict__ Wk, const float* __restrict__ bk,
    const float* __restrict__ Wv, const float* __restrict__ bv,
    unsigned short* __restrict__ qs, unsigned short* __restrict__ ks,
    unsigned short* __restrict__ vs)
{
    // padded stride: 18 uints = 36 bf16 per row (rows hold 32 bf16 of data)
    __shared__ __align__(16) unsigned lA[128 * 18];
    __shared__ __align__(16) unsigned lB[128 * 18];

    const int tid  = threadIdx.x;
    const int wave = tid >> 5, lane = tid & 31;
    const int half = lane >> 4, l16 = lane & 15;
    const int waveM = wave >> 2;        // 0..1 -> 64-row band
    const int waveN = wave & 3;         // 0..3 -> 32-col band

    const int tileN = blockIdx.x * 128;     // 0..3071 (concat Q|K|V)
    const int tileM = blockIdx.y * 128;     // 0..4095 (= b*512 + s)
    const int mat   = tileN >> 10;
    const int nOff  = tileN & 1023;
    const float* W            = (mat == 0) ? Wq : (mat == 1) ? Wk : Wv;
    const float* bias         = (mat == 0) ? bq : (mat == 1) ? bk : bv;
    unsigned short* outp      = (mat == 0) ? qs : (mat == 1) ? ks : vs;

    floatx8 acc[4][2];
    #pragma unroll
    for (int i = 0; i < 4; ++i)
        #pragma unroll
        for (int n = 0; n < 2; ++n) acc[i][n] = floatx8{};

    for (int kk = 0; kk < Dd; kk += 32) {
        __syncthreads();
        // stage X tile 128x32 (f32 -> bf16), vector float4 loads
        #pragma unroll
        for (int it = 0; it < 4; ++it) {
            int idx = tid + it * 256;            // 0..1023 float4 units
            int row = idx >> 3, c4 = idx & 7;
            float4 xv = *(const float4*)(X + (size_t)(tileM + row) * Dd + kk + c4 * 4);
            lA[row * 18 + c4 * 2 + 0] = pack2bf(xv.x, xv.y);
            lA[row * 18 + c4 * 2 + 1] = pack2bf(xv.z, xv.w);
        }
        // stage W tile 128x32 (rows = output features, K-major like X)
        #pragma unroll
        for (int it = 0; it < 4; ++it) {
            int idx = tid + it * 256;
            int row = idx >> 3, c4 = idx & 7;
            float4 wv = *(const float4*)(W + (size_t)(nOff + row) * Dd + kk + c4 * 4);
            lB[row * 18 + c4 * 2 + 0] = pack2bf(wv.x, wv.y);
            lB[row * 18 + c4 * 2 + 1] = pack2bf(wv.z, wv.w);
        }
        __syncthreads();

        // B fragments (32x16): lane = N column, K = {half*8+2j | 16+half*8+2(j-4)}
        FragA bf[2];
        #pragma unroll
        for (int nt = 0; nt < 2; ++nt) {
            const unsigned* rp = &lB[(waveN * 32 + nt * 16 + l16) * 18 + half * 4];
            #pragma unroll
            for (int j = 0; j < 8; ++j) bf[nt].u[j] = rp[(j & 3) + (j >> 2) * 8];
        }
        #pragma unroll
        for (int mt = 0; mt < 4; ++mt) {
            // A fragment (16x32): lane = M%16; K = {half*8+2j | 16+half*8+2(j-4)}
            FragA af;
            const unsigned* rp = &lA[(waveM * 64 + mt * 16 + l16) * 18];
            af.u[0] = rp[half * 4 + 0]; af.u[1] = rp[half * 4 + 1];
            af.u[2] = rp[half * 4 + 2]; af.u[3] = rp[half * 4 + 3];
            af.u[4] = rp[8 + half * 4 + 0]; af.u[5] = rp[8 + half * 4 + 1];
            af.u[6] = rp[8 + half * 4 + 2]; af.u[7] = rp[8 + half * 4 + 3];
            #pragma unroll
            for (int nt = 0; nt < 2; ++nt)
                acc[mt][nt] = __builtin_amdgcn_wmma_f32_16x16x32_bf16(
                    false, af.v, false, bf[nt].v, (short)0, acc[mt][nt], false, false);
        }
    }

    // epilogue: +bias, write bf16 to [B][H][S][HD]
    #pragma unroll
    for (int nt = 0; nt < 2; ++nt) {
        const int   eGlob = nOff + waveN * 32 + nt * 16 + l16;  // feature in mat
        const float bval  = bias[eGlob];
        const int   h = eGlob >> 6, hd = eGlob & 63;
        #pragma unroll
        for (int mt = 0; mt < 4; ++mt) {
            #pragma unroll
            for (int r = 0; r < 8; ++r) {
                int mGlob = tileM + waveM * 64 + mt * 16 + r + half * 8;
                int b = mGlob >> 9, s = mGlob & 511;
                outp[(((size_t)(b * Hh + h)) * Ss + s) * HDd + hd] =
                    f2bf(acc[mt][nt][r] + bval);
            }
        }
    }
}

// ---------------------------------------------------------------------------
// Kernel 2: flash attention per (b, h, 64-row Q block).
// 128 threads / 4 waves; each wave owns 16 Q rows; loop over 8 K/V blocks.
// Q/K tiles staged with GLOBAL_LOAD_ASYNC_TO_LDS_B128 (ASYNCcnt path).
// ---------------------------------------------------------------------------
__global__ __launch_bounds__(128) void attn_kernel(
    const unsigned short* __restrict__ qs,
    const unsigned short* __restrict__ ks,
    const unsigned short* __restrict__ vs,
    const float* __restrict__ mask,     // [B,1,1,S]
    float* __restrict__ out)            // [B,S,D]
{
    // stride 36 uints = 72 bf16 per row (rows hold 64 bf16 of data)
    __shared__ __align__(16) unsigned       lQ[64 * 36];
    __shared__ __align__(16) unsigned       lK[64 * 36];
    __shared__ __align__(16) unsigned short lVt[64 * 72];       // [hd][row]
    __shared__ __align__(16) unsigned short lP[4 * 16 * 72];    // per-wave P

    const int tid  = threadIdx.x;
    const int wave = tid >> 5, lane = tid & 31;
    const int half = lane >> 4, l16 = lane & 15;
    const int qb = blockIdx.x, h = blockIdx.y, b = blockIdx.z;
    const size_t bh = ((size_t)b * Hh + h) * Ss * HDd;

    // stage Q tile once: async global->LDS (bf16, row-major, padded)
    {
        const unsigned short* qbase = qs + bh + (size_t)qb * 64 * HDd;
        #pragma unroll
        for (int it = 0; it < 4; ++it) {
            int idx = tid + it * 128;        // 0..511 b128 units (8 bf16 each)
            int row = idx >> 3, c = idx & 7;
            async_copy_b128(&lQ[row * 36 + c * 4], qbase + row * 64 + c * 8);
        }
    }
    wait_async();
    __syncthreads();

    // Q fragments (held in registers): rows wave*16..+15, K-steps d=0..31, 32..63
    FragA aq[2];
    #pragma unroll
    for (int ksi = 0; ksi < 2; ++ksi) {
        const unsigned* rp = &lQ[(wave * 16 + l16) * 36 + ksi * 16];
        aq[ksi].u[0] = rp[half * 4 + 0]; aq[ksi].u[1] = rp[half * 4 + 1];
        aq[ksi].u[2] = rp[half * 4 + 2]; aq[ksi].u[3] = rp[half * 4 + 3];
        aq[ksi].u[4] = rp[8 + half * 4 + 0]; aq[ksi].u[5] = rp[8 + half * 4 + 1];
        aq[ksi].u[6] = rp[8 + half * 4 + 2]; aq[ksi].u[7] = rp[8 + half * 4 + 3];
    }

    floatx8 oacc[4];
    #pragma unroll
    for (int i = 0; i < 4; ++i) oacc[i] = floatx8{};
    float m_run[8], l_run[8];
    #pragma unroll
    for (int r = 0; r < 8; ++r) { m_run[r] = -1e30f; l_run[r] = 0.0f; }

    const float scale = 0.125f;   // 1/sqrt(64)

    for (int j = 0; j < 8; ++j) {
        __syncthreads();
        // stage K block row-major via async copy (K-major for B frags of Q*K^T)
        const unsigned short* kb = ks + bh + (size_t)j * 64 * HDd;
        #pragma unroll
        for (int it = 0; it < 4; ++it) {
            int idx = tid + it * 128;
            int row = idx >> 3, c = idx & 7;
            async_copy_b128(&lK[row * 36 + c * 4], kb + row * 64 + c * 8);
        }
        // stage V block transposed: lVt[hd][row]  (K-major for B frags of P*V)
        const unsigned short* vb = vs + bh + (size_t)j * 64 * HDd;
        #pragma unroll
        for (int it = 0; it < 16; ++it) {
            int idx = tid + it * 128;        // 0..2047 uint units (2 bf16 each)
            int row = idx >> 5, cu = idx & 31;
            unsigned val = *(const unsigned*)(vb + row * 64 + cu * 2);
            lVt[(cu * 2 + 0) * 72 + row] = (unsigned short)(val & 0xffffu);
            lVt[(cu * 2 + 1) * 72 + row] = (unsigned short)(val >> 16);
        }
        wait_async();
        __syncthreads();

        // S = Q * K^T   (4 N-tiles of 16 key rows, 2 K-steps over d)
        floatx8 sacc[4];
        #pragma unroll
        for (int nt = 0; nt < 4; ++nt) sacc[nt] = floatx8{};
        #pragma unroll
        for (int nt = 0; nt < 4; ++nt) {
            #pragma unroll
            for (int ksi = 0; ksi < 2; ++ksi) {
                FragA bf;
                const unsigned* rp = &lK[(nt * 16 + l16) * 36 + ksi * 16 + half * 4];
                #pragma unroll
                for (int jj = 0; jj < 8; ++jj) bf.u[jj] = rp[(jj & 3) + (jj >> 2) * 8];
                sacc[nt] = __builtin_amdgcn_wmma_f32_16x16x32_bf16(
                    false, aq[ksi].v, false, bf.v, (short)0, sacc[nt], false, false);
            }
        }

        // online softmax (rows r+8*half; 16-lane shuffle reductions per half)
        float mv[4];
        #pragma unroll
        for (int nt = 0; nt < 4; ++nt) mv[nt] = mask[b * Ss + j * 64 + nt * 16 + l16];
        float alpha[8];
        #pragma unroll
        for (int r = 0; r < 8; ++r) {
            float s0 = sacc[0][r] * scale + mv[0];
            float s1 = sacc[1][r] * scale + mv[1];
            float s2 = sacc[2][r] * scale + mv[2];
            float s3 = sacc[3][r] * scale + mv[3];
            float vmax = hmax16(fmaxf(fmaxf(s0, s1), fmaxf(s2, s3)));
            float mnew = fmaxf(m_run[r], vmax);
            float a    = __expf(m_run[r] - mnew);
            float p0 = __expf(s0 - mnew), p1 = __expf(s1 - mnew);
            float p2 = __expf(s2 - mnew), p3 = __expf(s3 - mnew);
            float rs = hsum16(p0 + p1 + p2 + p3);
            l_run[r] = l_run[r] * a + rs;
            m_run[r] = mnew;
            alpha[r] = a;
            sacc[0][r] = p0; sacc[1][r] = p1; sacc[2][r] = p2; sacc[3][r] = p3;
        }
        #pragma unroll
        for (int nt = 0; nt < 4; ++nt)
            #pragma unroll
            for (int r = 0; r < 8; ++r) oacc[nt][r] *= alpha[r];

        // spill P (C layout) -> LDS as bf16 in A-source layout (row-major per wave)
        unsigned short* lPw = lP + wave * 16 * 72;
        #pragma unroll
        for (int nt = 0; nt < 4; ++nt)
            #pragma unroll
            for (int r = 0; r < 8; ++r)
                lPw[(r + half * 8) * 72 + nt * 16 + l16] = f2bf(sacc[nt][r]);
        asm volatile("s_wait_dscnt 0" ::: "memory");   // wave-local LDS RAW fence

        // O += P * V
        const unsigned* lPu = (const unsigned*)lP;
        #pragma unroll
        for (int ksi = 0; ksi < 2; ++ksi) {
            FragA pa;
            const unsigned* rp = lPu + (wave * 16 + l16) * 36 + ksi * 16;
            pa.u[0] = rp[half * 4 + 0]; pa.u[1] = rp[half * 4 + 1];
            pa.u[2] = rp[half * 4 + 2]; pa.u[3] = rp[half * 4 + 3];
            pa.u[4] = rp[8 + half * 4 + 0]; pa.u[5] = rp[8 + half * 4 + 1];
            pa.u[6] = rp[8 + half * 4 + 2]; pa.u[7] = rp[8 + half * 4 + 3];
            #pragma unroll
            for (int nt2 = 0; nt2 < 4; ++nt2) {
                FragA bvf;
                const unsigned* vp = (const unsigned*)lVt + (nt2 * 16 + l16) * 36 + ksi * 16 + half * 4;
                #pragma unroll
                for (int jj = 0; jj < 8; ++jj) bvf.u[jj] = vp[(jj & 3) + (jj >> 2) * 8];
                oacc[nt2] = __builtin_amdgcn_wmma_f32_16x16x32_bf16(
                    false, pa.v, false, bvf.v, (short)0, oacc[nt2], false, false);
            }
        }
    }

    // epilogue: normalize and scatter to [B,S,D] fp32
    #pragma unroll
    for (int nt = 0; nt < 4; ++nt) {
        #pragma unroll
        for (int r = 0; r < 8; ++r) {
            int m  = r + half * 8;
            int s  = qb * 64 + wave * 16 + m;
            int hd = nt * 16 + l16;
            out[((size_t)(b * Ss + s)) * Dd + h * HDd + hd] = oacc[nt][r] / l_run[r];
        }
    }
}

extern "C" void kernel_launch(void* const* d_in, const int* in_sizes, int n_in,
                              void* d_out, int out_size, void* d_ws, size_t ws_size,
                              hipStream_t stream) {
    const float* X    = (const float*)d_in[0];
    const float* mask = (const float*)d_in[1];
    const float* Wq   = (const float*)d_in[2];
    const float* bq   = (const float*)d_in[3];
    const float* Wk   = (const float*)d_in[4];
    const float* bk   = (const float*)d_in[5];
    const float* Wv   = (const float*)d_in[6];
    const float* bv   = (const float*)d_in[7];
    float* out = (float*)d_out;

    const size_t perMat = (size_t)Bb * Hh * Ss * HDd;   // 4M bf16 = 8 MB each
    unsigned short* qsc = (unsigned short*)d_ws;
    unsigned short* ksc = qsc + perMat;
    unsigned short* vsc = ksc + perMat;

    dim3 g1(3 * Dd / 128, (Bb * Ss) / 128);   // 24 x 32 workgroups
    qkv_gemm_kernel<<<g1, 256, 0, stream>>>(X, Wq, bq, Wk, bk, Wv, bv, qsc, ksc, vsc);

    dim3 g2(Ss / 64, Hh, Bb);                 // 8 x 16 x 8 workgroups
    attn_kernel<<<g2, 128, 0, stream>>>(qsc, ksc, vsc, mask, out);
}
